// CausalSelfAttention_8856222564388
// MI455X (gfx1250) — matrix operand order repaired and checked by reading, not compiled, not run on hardware
//
#include <hip/hip_runtime.h>
#include <hip/hip_bf16.h>

// CDNA5 / gfx1250: wave32, WMMA bf16 (16x16x32, f32 accumulate),
// TDM (tensor_load_to_lds) double-buffered GEMM.

typedef __attribute__((ext_vector_type(16))) __bf16       v16bf;
typedef __attribute__((ext_vector_type(8)))  float        v8f;
typedef __attribute__((ext_vector_type(4)))  unsigned int u32x4;
typedef __attribute__((ext_vector_type(8)))  int          i32x8;
typedef __attribute__((ext_vector_type(4)))  int          i32x4;

union Frag16 { v16bf v; u32x4 q[2]; unsigned short u[16]; };

#define TT 2048
#define CC 1024
#define HH 16
#define DD 64

__device__ __forceinline__ unsigned short f2bf(float f) {
  unsigned u = __builtin_bit_cast(unsigned, f);
  unsigned r = (u + 0x7FFFu + ((u >> 16) & 1u)) >> 16;  // round-nearest-even
  return (unsigned short)r;
}

// Low 32 bits of a flat pointer into LDS == byte offset in the group segment
// (LDS aperture encodes the segment in addr[63:32]).
__device__ __forceinline__ unsigned lds_byte_off(const void* p) {
  return (unsigned)(unsigned long long)(p);
}

// Issue one TDM load of a [tileRows x 64]-element bf16 tile from a row-major
// [rows x CC] tensor into LDS (row-major, 64 elements per row).
__device__ __forceinline__ void tdm_tile_load(const unsigned short* g,
                                              unsigned ldsOff,
                                              unsigned tensorRows) {
  const unsigned long long ga = (unsigned long long)g;
  u32x4 g0;
  g0[0] = 1u;                                            // count = 1 descriptor
  g0[1] = ldsOff;                                        // lds_addr (bytes)
  g0[2] = (unsigned)ga;                                  // global_addr[31:0]
  g0[3] = ((unsigned)(ga >> 32) & 0x01FFFFFFu)           // global_addr[56:32]
        | 0x80000000u;                                   // type = 2 ("image")

  const unsigned td0 = CC;        // tensor_dim0 (K extent, elements)
  const unsigned td1 = tensorRows;// tensor_dim1 (row extent)
  const unsigned tile0 = 64;      // tile K extent  (128B rows)
  const unsigned tile1 = 128;     // tile row extent
  const unsigned s0 = CC;         // tensor_dim0_stride (elements)

  i32x8 g1;
  g1[0] = 0x00010000;                                    // data_size=1 (2B)
  g1[1] = (int)((td0 & 0xFFFFu) << 16);                  // [63:48]=td0 lo
  g1[2] = (int)((td0 >> 16) | ((td1 & 0xFFFFu) << 16));  // td0 hi | td1 lo
  g1[3] = (int)((td1 >> 16) | (tile0 << 16));            // td1 hi | tile_dim0
  g1[4] = (int)(tile1);                                  // tile_dim1 | tile_dim2=0
  g1[5] = (int)(s0);                                     // stride0[31:0]
  g1[6] = 0;                                             // stride0 hi | stride1 lo
  g1[7] = 0;                                             // stride1 hi
  i32x4 z4 = {0, 0, 0, 0};
#if __clang_major__ >= 23
  i32x8 z8 = {0, 0, 0, 0, 0, 0, 0, 0};
  __builtin_amdgcn_tensor_load_to_lds(g0, g1, z4, z4, z8, 0);
#else
  __builtin_amdgcn_tensor_load_to_lds(g0, g1, z4, z4, 0);
#endif
}

// ---------------------------------------------------------------- conversions

__global__ __launch_bounds__(256) void f32_to_bf16_k(
    const float* __restrict__ src, unsigned short* __restrict__ dst, int n) {
  int i = blockIdx.x * 256 + threadIdx.x;
  if (i < n) dst[i] = f2bf(src[i]);
}

// W stored [K=in][N=out]; produce Wt[N][K] bf16 so B-fragments are contiguous-K.
__global__ __launch_bounds__(256) void wconv_k(
    const float* __restrict__ W, unsigned short* __restrict__ Wt) {
  int idx = blockIdx.x * 256 + threadIdx.x;      // 0 .. CC*CC-1
  int k = idx >> 10, n = idx & (CC - 1);
  Wt[(size_t)n * CC + k] = f2bf(W[idx]);
}

// ---------------------------------------------------------------- GEMM
// C[M=8192, N=1024] = A[M,K=1024] * Bt[N,K]^T + bias
// Block tile 128x128, 8 waves of 32x64, TDM double-buffered LDS staging.
// mode 0: store bf16 -> [B,H,T,D]   (q, k)
// mode 1: store bf16 -> [B,H,D,T]   (v transposed for P*V fragments)
// mode 2: store f32  -> [M,N]       (final projection)
__global__ __launch_bounds__(256) void gemm_bf16_k(
    const unsigned short* __restrict__ A,
    const unsigned short* __restrict__ Bt,
    const float* __restrict__ bias,
    void* __restrict__ dst, int mode) {
  __shared__ __align__(16) unsigned short Al[2][128 * 64];
  __shared__ __align__(16) unsigned short Bl[2][128 * 64];

  const int lane = threadIdx.x & 31;
  const int wid  = threadIdx.x >> 5;
  const int lm = lane & 15, hi = lane >> 4, kbo = hi * 8;
  const int mg = (wid >> 1) * 32;          // wave row group within block
  const int ng = (wid & 1) * 64;           // wave col group within block
  const int mblk = blockIdx.y * 128;
  const int nblk = blockIdx.x * 128;

  v8f acc[2][4];
#pragma unroll
  for (int rt = 0; rt < 2; ++rt)
#pragma unroll
    for (int nt = 0; nt < 4; ++nt)
#pragma unroll
      for (int e = 0; e < 8; ++e) acc[rt][nt][e] = 0.f;

  if (wid == 0) {                           // prologue: stage 0
    tdm_tile_load(A  + (size_t)mblk * CC, lds_byte_off(&Al[0][0]), 8192);
    tdm_tile_load(Bt + (size_t)nblk * CC, lds_byte_off(&Bl[0][0]), CC);
  }

  const int NIT = CC / 64;                  // 16 K-stages
  for (int it = 0; it < NIT; ++it) {
    const int s = it & 1;
    if (wid == 0) {
      if (it + 1 < NIT) {
        const int kk = (it + 1) * 64;
        tdm_tile_load(A  + (size_t)mblk * CC + kk, lds_byte_off(&Al[s ^ 1][0]), 8192);
        tdm_tile_load(Bt + (size_t)nblk * CC + kk, lds_byte_off(&Bl[s ^ 1][0]), CC);
        __builtin_amdgcn_s_wait_tensorcnt((short)2);   // stage `it` landed
      } else {
        __builtin_amdgcn_s_wait_tensorcnt((short)0);
      }
    }
    __syncthreads();
#pragma unroll
    for (int kk2 = 0; kk2 < 2; ++kk2) {
      Frag16 a[2];
#pragma unroll
      for (int rt = 0; rt < 2; ++rt) {     // A 16x32 frags: lane=row, split K
        const unsigned short* ap =
            &Al[s][(mg + rt * 16 + lm) * 64 + kk2 * 32 + kbo];
        a[rt].q[0] = *(const u32x4*)ap;
        a[rt].q[1] = *(const u32x4*)(ap + 16);
      }
#pragma unroll
      for (int nt = 0; nt < 4; ++nt) {     // B 32x16 frags: lane=col, split K
        const unsigned short* bp =
            &Bl[s][(ng + nt * 16 + lm) * 64 + kk2 * 32 + hi * 8];
        Frag16 b;
        b.q[0] = *(const u32x4*)bp;
        b.q[1] = *(const u32x4*)(bp + 16);
#pragma unroll
        for (int rt = 0; rt < 2; ++rt)
          acc[rt][nt] = __builtin_amdgcn_wmma_f32_16x16x32_bf16(
              false, a[rt].v, false, b.v, (short)0, acc[rt][nt], false, false);
      }
    }
    __syncthreads();
  }

#pragma unroll
  for (int rt = 0; rt < 2; ++rt) {
#pragma unroll
    for (int nt = 0; nt < 4; ++nt) {
      const int col = nblk + ng + nt * 16 + lm;
      const float bv = bias[col];
#pragma unroll
      for (int r = 0; r < 8; ++r) {        // C/D: vgpr r -> row r+8*hi
        const int M = mblk + mg + rt * 16 + r + hi * 8;
        const float val = acc[rt][nt][r] + bv;
        if (mode == 2) {
          ((float*)dst)[(size_t)M * CC + col] = val;
        } else {
          const int bi = M >> 11, t = M & (TT - 1);
          const int hh = col >> 6, d = col & (DD - 1);
          const size_t idx = (mode == 0)
              ? ((size_t)(bi * HH + hh) * TT + t) * DD + d
              : ((size_t)(bi * HH + hh) * DD + d) * TT + t;
          ((unsigned short*)dst)[idx] = f2bf(val);
        }
      }
    }
  }
}

// ---------------------------------------------------------------- attention
// Flash-style causal attention. Block = 64 query rows (4 waves x 16 rows),
// key blocks of 32. q,k: [B,H,T,D] bf16; vt: [B,H,D,T] bf16; y: [B,T,C] bf16.
__global__ __launch_bounds__(128) void attn_k(
    const unsigned short* __restrict__ q,
    const unsigned short* __restrict__ kmat,
    const unsigned short* __restrict__ vt,
    unsigned short* __restrict__ y) {
  const int lane = threadIdx.x & 31, wid = threadIdx.x >> 5;
  const int lm = lane & 15, hi = lane >> 4, kbo = hi * 8;
  const int i0 = blockIdx.x * 64;
  const int bh = blockIdx.y;                 // b*16 + h
  const int b = bh >> 4, h = bh & (HH - 1);
  const size_t bhT = (size_t)bh * TT;
  const int rbase = i0 + wid * 16;

  __shared__ unsigned short Plds[4][16][32];  // per-wave P staging (C->A layout)

  Frag16 qf[2];                               // Q row tile, K=0..63 in 2 frags
  {
    const unsigned short* qrow = q + (bhT + rbase + lm) * DD;
#pragma unroll
    for (int kk2 = 0; kk2 < 2; ++kk2) {
      qf[kk2].q[0] = *(const u32x4*)(qrow + kk2 * 32 + kbo);
      qf[kk2].q[1] = *(const u32x4*)(qrow + kk2 * 32 + 16 + kbo);
    }
  }

  float mrun[8], lrun[8];
  v8f o[4];
#pragma unroll
  for (int r = 0; r < 8; ++r) { mrun[r] = -__builtin_inff(); lrun[r] = 0.f; }
#pragma unroll
  for (int nt = 0; nt < 4; ++nt)
#pragma unroll
    for (int e = 0; e < 8; ++e) o[nt][e] = 0.f;

  for (int j0 = 0; j0 < i0 + 64; j0 += 32) {
    // ---- S = Q * K^T  (two 16-col tiles, K-dim = D = 64)
    v8f s[2];
#pragma unroll
    for (int ct = 0; ct < 2; ++ct) {
#pragma unroll
      for (int e = 0; e < 8; ++e) s[ct][e] = 0.f;
#pragma unroll
      for (int kk2 = 0; kk2 < 2; ++kk2) {
        const unsigned short* krow =
            kmat + (bhT + j0 + ct * 16 + lm) * DD + kk2 * 32 + hi * 8;
        Frag16 kf;
        kf.q[0] = ((const u32x4*)krow)[0];
        kf.q[1] = ((const u32x4*)krow)[2];
        s[ct] = __builtin_amdgcn_wmma_f32_16x16x32_bf16(
            false, qf[kk2].v, false, kf.v, (short)0, s[ct], false, false);
      }
    }
    // ---- scale, causal mask, online softmax (row stats in 16-lane halves)
#pragma unroll
    for (int r = 0; r < 8; ++r) {
      const int grow = rbase + r + hi * 8;
      float v0 = s[0][r] * 0.125f;
      float v1 = s[1][r] * 0.125f;
      if (j0 + lm > grow)      v0 = -__builtin_inff();
      if (j0 + 16 + lm > grow) v1 = -__builtin_inff();
      float mx = fmaxf(v0, v1);
      mx = fmaxf(mx, __shfl_xor(mx, 1, 32));
      mx = fmaxf(mx, __shfl_xor(mx, 2, 32));
      mx = fmaxf(mx, __shfl_xor(mx, 4, 32));
      mx = fmaxf(mx, __shfl_xor(mx, 8, 32));
      const float mn = fmaxf(mrun[r], mx);
      const float alpha = __expf(mrun[r] - mn);
      const float p0 = __expf(v0 - mn);
      const float p1 = __expf(v1 - mn);
      float rs = p0 + p1;
      rs += __shfl_xor(rs, 1, 32);
      rs += __shfl_xor(rs, 2, 32);
      rs += __shfl_xor(rs, 4, 32);
      rs += __shfl_xor(rs, 8, 32);
      lrun[r] = lrun[r] * alpha + rs;
      mrun[r] = mn;
#pragma unroll
      for (int nt = 0; nt < 4; ++nt) o[nt][r] *= alpha;
      const int row = r + hi * 8;
      Plds[wid][row][lm]      = f2bf(p0);
      Plds[wid][row][16 + lm] = f2bf(p1);
    }
    __syncthreads();
    // ---- reload P in A-layout, O += P * V
    Frag16 pf;
    const unsigned short* pr = &Plds[wid][lm][0];
    pf.q[0] = *(const u32x4*)(pr + kbo);
    pf.q[1] = *(const u32x4*)(pr + 16 + kbo);
#pragma unroll
    for (int nt = 0; nt < 4; ++nt) {
      const unsigned short* vrow =
          vt + ((size_t)bh * DD + nt * 16 + lm) * TT + j0 + hi * 8;
      Frag16 vf;
      vf.q[0] = ((const u32x4*)vrow)[0];
      vf.q[1] = ((const u32x4*)vrow)[2];
      o[nt] = __builtin_amdgcn_wmma_f32_16x16x32_bf16(
          false, pf.v, false, vf.v, (short)0, o[nt], false, false);
    }
    __syncthreads();
  }
  // ---- normalize and store y[B,T,C] bf16
#pragma unroll
  for (int r = 0; r < 8; ++r) {
    const float inv = 1.f / lrun[r];
    const int t = rbase + r + hi * 8;
#pragma unroll
    for (int nt = 0; nt < 4; ++nt) {
      const int d = nt * 16 + lm;
      y[((size_t)b * TT + t) * CC + h * DD + d] = f2bf(o[nt][r] * inv);
    }
  }
}

// ---------------------------------------------------------------- launcher

extern "C" void kernel_launch(void* const* d_in, const int* in_sizes, int n_in,
                              void* d_out, int out_size, void* d_ws, size_t ws_size,
                              hipStream_t stream) {
  (void)in_sizes; (void)n_in; (void)out_size; (void)ws_size;
  const float* x  = (const float*)d_in[0];
  const float* Wq = (const float*)d_in[1];
  const float* bq = (const float*)d_in[2];
  const float* Wk = (const float*)d_in[3];
  const float* bk = (const float*)d_in[4];
  const float* Wv = (const float*)d_in[5];
  const float* bv = (const float*)d_in[6];
  const float* Wp = (const float*)d_in[7];
  const float* bp = (const float*)d_in[8];

  const int Bb = 4;
  const int M  = Bb * TT;                 // 8192
  char* ws = (char*)d_ws;
  size_t off = 0;
  unsigned short* xb  = (unsigned short*)(ws + off); off += (size_t)M * CC * 2;
  unsigned short* wqt = (unsigned short*)(ws + off); off += (size_t)CC * CC * 2;
  unsigned short* wkt = (unsigned short*)(ws + off); off += (size_t)CC * CC * 2;
  unsigned short* wvt = (unsigned short*)(ws + off); off += (size_t)CC * CC * 2;
  unsigned short* wpt = (unsigned short*)(ws + off); off += (size_t)CC * CC * 2;
  unsigned short* qb  = (unsigned short*)(ws + off); off += (size_t)M * CC * 2;
  unsigned short* kb  = (unsigned short*)(ws + off); off += (size_t)M * CC * 2;
  unsigned short* vtb = (unsigned short*)(ws + off); off += (size_t)M * CC * 2;
  unsigned short* yb  = (unsigned short*)(ws + off); off += (size_t)M * CC * 2;

  const int nx = M * CC;
  f32_to_bf16_k<<<(nx + 255) / 256, 256, 0, stream>>>(x, xb, nx);
  wconv_k<<<(CC * CC) / 256, 256, 0, stream>>>(Wq, wqt);
  wconv_k<<<(CC * CC) / 256, 256, 0, stream>>>(Wk, wkt);
  wconv_k<<<(CC * CC) / 256, 256, 0, stream>>>(Wv, wvt);
  wconv_k<<<(CC * CC) / 256, 256, 0, stream>>>(Wp, wpt);

  dim3 gg(CC / 128, M / 128);             // (8, 64)
  gemm_bf16_k<<<gg, 256, 0, stream>>>(xb, wqt, bq, qb, 0);
  gemm_bf16_k<<<gg, 256, 0, stream>>>(xb, wkt, bk, kb, 0);
  gemm_bf16_k<<<gg, 256, 0, stream>>>(xb, wvt, bv, vtb, 1);

  attn_k<<<dim3(TT / 64, Bb * HH), 128, 0, stream>>>(qb, kb, vtb, yb);

  gemm_bf16_k<<<gg, 256, 0, stream>>>(yb, wpt, bp, d_out, 2);
}
